// GTRNN_81363860455486
// MI455X (gfx1250) — matrix operand order repaired and checked by reading, not compiled, not run on hardware
//
#include <hip/hip_runtime.h>

// ---------------------------------------------------------------------------
// 3-layer RNN scan for MI455X (gfx1250).
// Latency-bound problem -> 2 persistent workgroups (one per 16-row batch
// tile), each filling one WGP with 32 wave32 waves; each wave owns a 16-col
// N-chunk of the 512 outputs. Activations resident in LDS (116 KB of the
// 320 KB WGP LDS, double-buffered h). Weights packed once to bf16 WMMA-B
// tiles and streamed from L2 every step (4.5 MB working set, hot in the
// 192 MB L2) as GLOBAL loads -- an opaque per-step offset blocks LICM from
// hoisting/spilling them while preserving the global address space (so the
// stream uses LOADcnt only, decoupled from the LDS DScnt path).
// Matrix math: v_wmma_f32_16x16x32_bf16 with f32 accumulation; tanh (TRANS
// op) co-executes with the XDL WMMAs.
// ---------------------------------------------------------------------------

typedef __attribute__((ext_vector_type(16))) __bf16 v16bf;
typedef __attribute__((ext_vector_type(8)))  __bf16 v8bf;
typedef __attribute__((ext_vector_type(8)))  float  v8f;

constexpr int kB = 32, kT = 1024, kI = 512, kH = 512, kL = 3;
constexpr int kRows = 16;            // M tile (16 batch rows per workgroup)
constexpr int kStride = kH + 8;      // LDS row stride in bf16 (bank-conflict pad)
constexpr int kTileElems = kRows * kStride;
constexpr int kSmemElems = (1 + 2 * kL) * kTileElems;   // y + double-buffered h[L]
constexpr int kSmemBytes = kSmemElems * 2;              // 116,480 B < 320 KB WGP LDS

// ---------------------------------------------------------------------------
// Pack f32 [out,in] weights into bf16 WMMA-B tiles.
// Block (n_tile, k_chunk) = 16 cols x 32 K, stored as 32 lanes x 16 bf16:
//   lane L holds column n0+(L&15), K values k0+(L>>4)*16 .. +15 (contiguous).
// GEMM loop then reads one aligned 32 B chunk per lane per wmma (a coalesced
// 1 KB per wave) matching the contiguous k order of the A fragments.
// ---------------------------------------------------------------------------
__global__ void pack_weights(const float* __restrict__ Wxh,
                             const float* __restrict__ Whh,
                             const float* __restrict__ Why,
                             __bf16* __restrict__ wp) {
  int idx = blockIdx.x * blockDim.x + threadIdx.x;     // 3 mats * 3 layers * 512*512
  int mat = idx / (kL * kH * kH);
  int rem = idx % (kL * kH * kH);
  const float* src = (mat == 0) ? Wxh : (mat == 1) ? Whh : Why;
  int l = rem / (kH * kH);
  int r2 = rem % (kH * kH);
  int n = r2 / kH;                                     // output row of W [out,in]
  int k = r2 % kH;
  float v = src[l * kH * kH + n * kH + k];
  int lane = ((k >> 4) & 1) * 16 + (n & 15);
  int e    = k & 15;
  long dst = (long)(mat * kL + l) * (kH * kH)          // matrix base
           + (long)(n >> 4) * (16 * kH)                // n-tile (32 per matrix)
           + (long)(k >> 5) * 512                      // k-chunk (16 per matrix)
           + lane * 16 + e;
  wp[dst] = (__bf16)v;
}

// A fragment (16x32 bf16) from LDS row-major buffer, in this file's contiguous k order:
// lane m (0-15): K 0..15 ; lane m+16: K 16..31 (contiguous, matches packed B).
__device__ __forceinline__ v16bf load_a_frag(const __bf16* base, int row,
                                             int khalf, int k0) {
  const __bf16* p = base + row * kStride + k0 + khalf * 16;
  union { v16bf v; v8bf h[2]; } u;
  u.h[0] = *(const v8bf*)(p);        // ds_load_b128
  u.h[1] = *(const v8bf*)(p + 8);    // ds_load_b128
  return u.v;
}

__device__ __forceinline__ v8f wmma_bf16(v16bf a, v16bf b, v8f c) {
  return __builtin_amdgcn_wmma_f32_16x16x32_bf16(false, a, false, b,
                                                 (short)0, c, false, false);
}

__global__ void __launch_bounds__(1024, 1)
rnn_persistent(const float* __restrict__ xs,    // [B,T,I]
               const float* __restrict__ hs0,   // [B,L,H]
               const float* __restrict__ b_hh,  // [L,H]
               const float* __restrict__ b_xh,  // [L,H]
               const float* __restrict__ b_hy,  // [L,H]
               const __bf16* __restrict__ wp,   // packed weights
               float* __restrict__ out) {       // [B,T,H] ++ [B,L,H]
  extern __shared__ __bf16 smem[];
  __bf16* ylds = smem;                                  // [16][kStride]
  // h double buffers: smem + (1 + buf*kL + l)*kTileElems

  const int tid   = (int)threadIdx.x;
  const int wave  = tid >> 5;
  const int lane  = tid & 31;
  const int col16 = lane & 15;       // A-row for loads, C/D-column for stores
  const int khalf = lane >> 4;
  const int nc    = wave * 16 + col16;   // this lane's output column
  const int mbase = (int)blockIdx.x * kRows;

  // Hoisted per-lane biases and weight base pointers (invariant over t).
  float bg[kL], by[kL];
  const __bf16 *wxh[kL], *whh[kL], *why[kL];
#pragma unroll
  for (int l = 0; l < kL; ++l) {
    bg[l] = b_xh[l * kH + nc] + b_hh[l * kH + nc];
    by[l] = b_hy[l * kH + nc];
    const long laneoff = (long)wave * (16 * kH) + lane * 16;
    wxh[l] = wp + (long)(0 * kL + l) * (kH * kH) + laneoff;
    whh[l] = wp + (long)(1 * kL + l) * (kH * kH) + laneoff;
    why[l] = wp + (long)(2 * kL + l) * (kH * kH) + laneoff;
  }

  // Initial hidden state -> h buffer 0 (t=0 reads buf t&1 == 0).
  for (int i = tid; i < kL * kRows * kH; i += 1024) {
    int l = i / (kRows * kH);
    int r = (i / kH) & 15;
    int c = i & (kH - 1);
    smem[(1 + 0 * kL + l) * kTileElems + r * kStride + c] =
        (__bf16)hs0[((long)(mbase + r) * kL + l) * kH + c];
  }
  __syncthreads();

  for (int t = 0; t < kT; ++t) {
    const int rb = t & 1, wb = rb ^ 1;

    // Stage x_t into ylds as bf16 (two global_load_b128 per thread).
    {
      int r  = tid >> 6;                 // 0..15
      int c0 = (tid & 63) * 8;           // 0..504
      const float4* xp =
          (const float4*)(xs + ((long)(mbase + r) * kT + t) * kI + c0);
      float4 x0 = xp[0], x1 = xp[1];
      __bf16* yp = ylds + r * kStride + c0;
      yp[0] = (__bf16)x0.x; yp[1] = (__bf16)x0.y;
      yp[2] = (__bf16)x0.z; yp[3] = (__bf16)x0.w;
      yp[4] = (__bf16)x1.x; yp[5] = (__bf16)x1.y;
      yp[6] = (__bf16)x1.z; yp[7] = (__bf16)x1.w;
    }
    __syncthreads();

    // Opaque zero, regenerated each timestep: adding it to the weight
    // pointers defeats LICM (no hoist -> no spill) but keeps the base
    // pointer's GLOBAL address space, so codegen stays global_load_b128.
    int jit = 0;
    asm volatile("" : "+v"(jit));

#pragma unroll
    for (int l = 0; l < kL; ++l) {
      const __bf16* hrd = smem + (1 + rb * kL + l) * kTileElems;
      __bf16*       hwr = smem + (1 + wb * kL + l) * kTileElems;

      const __bf16* pwx = wxh[l] + jit;
      const __bf16* pwh = whh[l] + jit;
      const __bf16* pwy = why[l] + jit;

      // gate = y @ Wxh^T + h_prev @ Whh^T  (K = 512, 16 chunks of 32),
      // with double-buffered weight fragments for load/WMMA overlap.
      v8f acc = {};
      v16bf bx = *(const v16bf*)(pwx);
      v16bf bh = *(const v16bf*)(pwh);
#pragma unroll
      for (int j = 0; j < 16; ++j) {
        v16bf bx_n, bh_n;
        if (j < 15) {
          bx_n = *(const v16bf*)(pwx + (j + 1) * 512);
          bh_n = *(const v16bf*)(pwh + (j + 1) * 512);
        }
        const int k0 = j * 32;
        v16bf ay = load_a_frag(ylds, col16, khalf, k0);
        acc = wmma_bf16(ay, bx, acc);
        v16bf ah = load_a_frag(hrd, col16, khalf, k0);
        acc = wmma_bf16(ah, bh, acc);
        bx = bx_n; bh = bh_n;
      }
      // h_new = tanh(gate + b); write buffer -> no hazard vs. h_prev readers.
#pragma unroll
      for (int r = 0; r < 8; ++r) {
        float hv = tanhf(acc[r] + bg[l]);
        hwr[(r + khalf * 8) * kStride + nc] = (__bf16)hv;
      }
      __syncthreads();   // all h_new tiles visible; all old-y reads retired

      // y = h_new @ Why^T + b (same double-buffered stream)
      v8f accy = {};
      v16bf bw = *(const v16bf*)(pwy);
#pragma unroll
      for (int j = 0; j < 16; ++j) {
        v16bf bw_n;
        if (j < 15) bw_n = *(const v16bf*)(pwy + (j + 1) * 512);
        v16bf ah = load_a_frag(hwr, col16, khalf, j * 32);
        accy = wmma_bf16(ah, bw, accy);
        bw = bw_n;
      }
      if (l < kL - 1) {
#pragma unroll
        for (int r = 0; r < 8; ++r)
          ylds[(r + khalf * 8) * kStride + nc] = (__bf16)(accy[r] + by[l]);
      } else {
#pragma unroll
        for (int r = 0; r < 8; ++r) {
          int m = mbase + r + khalf * 8;
          out[((long)m * kT + t) * kH + nc] = accy[r] + by[l];
        }
      }
      __syncthreads();   // ylds ready for next layer / next step's x-stage
    }
  }

  // Final hidden state lives in buffer (kT & 1) == 0.
  const long HS_OFF = (long)kB * kT * kH;
  for (int i = tid; i < kL * kRows * kH; i += 1024) {
    int l = i / (kRows * kH);
    int r = (i / kH) & 15;
    int c = i & (kH - 1);
    out[HS_OFF + ((long)(mbase + r) * kL + l) * kH + c] =
        (float)smem[(1 + 0 * kL + l) * kTileElems + r * kStride + c];
  }
}

extern "C" void kernel_launch(void* const* d_in, const int* in_sizes, int n_in,
                              void* d_out, int out_size, void* d_ws, size_t ws_size,
                              hipStream_t stream) {
  const float* xs  = (const float*)d_in[0];  // [B,T,I]
  const float* hs  = (const float*)d_in[1];  // [B,L,H]
  const float* Whh = (const float*)d_in[2];  // [L,H,H]
  const float* bhh = (const float*)d_in[3];  // [L,H]
  const float* Wxh = (const float*)d_in[4];  // [L,H,I]
  const float* bxh = (const float*)d_in[5];  // [L,H]
  const float* Why = (const float*)d_in[6];  // [L,H,H]
  const float* bhy = (const float*)d_in[7];  // [L,H]
  __bf16* wp = (__bf16*)d_ws;                // needs 3*3*512*512*2 = 4.5 MB
  float* out = (float*)d_out;

  // 1) Repack weights to bf16 WMMA tiles (runs every call; deterministic).
  const int total = 3 * kL * kH * kH;
  pack_weights<<<total / 256, 256, 0, stream>>>(Wxh, Whh, Why, wp);

  // 2) Persistent RNN: 2 workgroups (batch halves), 1024 threads = 32 waves,
  //    116 KB dynamic LDS (CDNA5 WGP supports 320 KB).
  (void)hipFuncSetAttribute((const void*)rnn_persistent,
                            hipFuncAttributeMaxDynamicSharedMemorySize,
                            kSmemBytes);
  rnn_persistent<<<2, 1024, kSmemBytes, stream>>>(xs, hs, bhh, bxh, bhy, wp, out);
}
